// DAInput_79001628443215
// MI455X (gfx1250) — matrix operand order repaired and checked by reading, not compiled, not run on hardware
//
#include <hip/hip_runtime.h>
#include <hip/hip_bf16.h>

// CDNA5 / gfx1250, wave32. WMMA bf16 16x16x32 tiles, bf16 activations end-to-end.

typedef __attribute__((ext_vector_type(16))) __bf16 v16bf;
typedef __attribute__((ext_vector_type(8)))  __bf16 v8bf;
typedef __attribute__((ext_vector_type(8)))  float  v8f;

#define NMAP 128
#define GEPS 1e-5f

__device__ __forceinline__ float wave_sum32(float x) {
#pragma unroll
  for (int o = 16; o; o >>= 1) x += __shfl_xor(x, o, 32);
  return x;
}

// ---------------------------------------------------------------------------
// Pre-pack one weight matrix W[K,128] (fp32) into WMMA B-fragment order, bf16.
// Fragment order (per ISA 7.12.2, 16-bit B 32x16):
//   lane holds col = ct*16 + (lane&15); elem e => K = ks*32 + 8*(lane>>4) + e + (e>=8 ? 8 : 0)
// Packed: Wp[((ks*8 + ct)*32 + lane)*16 + e]  -> B frag load = one 32B read.
// Launch: <<<K/32, 256>>>
// ---------------------------------------------------------------------------
__global__ __launch_bounds__(256)
void pack_w_kernel(const float* __restrict__ W, __bf16* __restrict__ Wp)
{
  const int tid  = blockIdx.x * 256 + threadIdx.x;
  const int lane = threadIdx.x & 31;
  const int ct   = (threadIdx.x >> 5) & 7;
  const int col  = ct * 16 + (lane & 15);
  const int krow = blockIdx.x * 32 + 8 * (lane >> 4);
  v16bf d;
#pragma unroll
  for (int e = 0; e < 16; ++e) d[e] = (__bf16)W[(size_t)(krow + e + (e >= 8 ? 8 : 0)) * NMAP + col];
  *(v16bf*)(Wp + (size_t)tid * 16) = d;
}

// ---------------------------------------------------------------------------
// Fused GEMM + GroupNorm(ng=1) + affine + optional residual + ReLU.
// C[N,128] = A[N,K] @ W[K,128]; A = A0 (bf16) for k<128, A1 (fp32) for k>=128.
// Workgroup: 32 rows x 128 cols; 8 waves; wave w owns cols [16w,16w+16),
// two 16-row tiles -> 2 wmma accumulators, KSTEPS unrolled wmma steps.
// A fragment (16-bit A 16x32): lane<16: row=lane, elems 0..7 => K+0..7,
//   8..15 => K+16..23; lane>=16: row=lane-16, K+8..15 / K+24..31.
// D fragment (f32 16x16): elem r => M = r + 8*(lane>>4), N = lane&15.
// ---------------------------------------------------------------------------
template <int KSTEPS>
__global__ __launch_bounds__(256)
void gemm_gn_kernel(const __bf16* __restrict__ A0, const float* __restrict__ A1,
                    const __bf16* __restrict__ Wp, const float* __restrict__ g,
                    const float* __restrict__ b,  const float* __restrict__ res,
                    float* __restrict__ out, __bf16* __restrict__ out_bf,
                    int do_relu)
{
  __shared__ float tile[32 * NMAP];  // 16 KB

  const int wave = threadIdx.x >> 5;
  const int lane = threadIdx.x & 31;
  const int hi   = lane >> 4;
  const int l15  = lane & 15;
  const int row0 = blockIdx.x * 32;
  const int col0 = wave * 16;

  v8f c0 = {}, c1 = {};
  const __bf16* wfrag = Wp + ((size_t)wave * 32 + lane) * 16;

#pragma unroll
  for (int ks = 0; ks < KSTEPS; ++ks) {
    const int kk = ks * 32;
    const v16bf bfrag = *(const v16bf*)(wfrag + (size_t)ks * (256 * 16));

#pragma unroll
    for (int rt = 0; rt < 2; ++rt) {
      const int row = row0 + rt * 16 + l15;
      v16bf a;
      if (kk < NMAP) {
        // bf16 A: two contiguous 16B chunks per fragment
        const __bf16* ap = A0 + (size_t)row * NMAP + kk + 8 * hi;
        const v8bf lo = *(const v8bf*)(ap);
        const v8bf hh = *(const v8bf*)(ap + 16);
        a = __builtin_shufflevector(lo, hh, 0, 1, 2, 3, 4, 5, 6, 7,
                                            8, 9, 10, 11, 12, 13, 14, 15);
      } else {
        // fp32 A (agg half of the concat), convert in-loader
        const float* ap = A1 + (size_t)row * NMAP + (kk - NMAP) + 8 * hi;
        const float4 q0 = *(const float4*)(ap);
        const float4 q1 = *(const float4*)(ap + 4);
        const float4 r0 = *(const float4*)(ap + 16);
        const float4 r1 = *(const float4*)(ap + 20);
        a[0]  = (__bf16)q0.x; a[1]  = (__bf16)q0.y; a[2]  = (__bf16)q0.z; a[3]  = (__bf16)q0.w;
        a[4]  = (__bf16)q1.x; a[5]  = (__bf16)q1.y; a[6]  = (__bf16)q1.z; a[7]  = (__bf16)q1.w;
        a[8]  = (__bf16)r0.x; a[9]  = (__bf16)r0.y; a[10] = (__bf16)r0.z; a[11] = (__bf16)r0.w;
        a[12] = (__bf16)r1.x; a[13] = (__bf16)r1.y; a[14] = (__bf16)r1.z; a[15] = (__bf16)r1.w;
      }
      if (rt == 0)
        c0 = __builtin_amdgcn_wmma_f32_16x16x32_bf16(false, a, false, bfrag,
                                                     (short)0, c0, false, false);
      else
        c1 = __builtin_amdgcn_wmma_f32_16x16x32_bf16(false, a, false, bfrag,
                                                     (short)0, c1, false, false);
    }
  }

  // spill accumulators to LDS for the cross-wave row reduction
#pragma unroll
  for (int r = 0; r < 8; ++r) {
    tile[(r + 8 * hi) * NMAP + col0 + l15]      = c0[r];
    tile[(16 + r + 8 * hi) * NMAP + col0 + l15] = c1[r];
  }
  __syncthreads();

  // fused GroupNorm (per-row over 128 ch) + affine (+res)(+relu) + store
#pragma unroll
  for (int rr = 0; rr < 4; ++rr) {
    const int row = wave * 4 + rr;
    const int col = lane * 4;
    const float4 vv = *(const float4*)&tile[row * NMAP + col];

    float s = vv.x + vv.y + vv.z + vv.w;
    float q = vv.x * vv.x + vv.y * vv.y + vv.z * vv.z + vv.w * vv.w;
    s = wave_sum32(s);
    q = wave_sum32(q);
    const float mu   = s * (1.0f / NMAP);
    const float rstd = rsqrtf(q * (1.0f / NMAP) - mu * mu + GEPS);

    const int gr = row0 + row;
    const float4 gg = *(const float4*)&g[col];
    const float4 bb = *(const float4*)&b[col];
    float o0 = (vv.x - mu) * rstd * gg.x + bb.x;
    float o1 = (vv.y - mu) * rstd * gg.y + bb.y;
    float o2 = (vv.z - mu) * rstd * gg.z + bb.z;
    float o3 = (vv.w - mu) * rstd * gg.w + bb.w;
    if (res) {
      const float4 rv = *(const float4*)&res[(size_t)gr * NMAP + col];
      o0 += rv.x; o1 += rv.y; o2 += rv.z; o3 += rv.w;
    }
    if (do_relu) {
      o0 = fmaxf(o0, 0.f); o1 = fmaxf(o1, 0.f);
      o2 = fmaxf(o2, 0.f); o3 = fmaxf(o3, 0.f);
    }
    if (out) *(float4*)&out[(size_t)gr * NMAP + col] = make_float4(o0, o1, o2, o3);
    if (out_bf) {
      __bf16* ob = out_bf + (size_t)gr * NMAP + col;
      ob[0] = (__bf16)o0; ob[1] = (__bf16)o1; ob[2] = (__bf16)o2; ob[3] = (__bf16)o3;
    }
  }
}

// ---------------------------------------------------------------------------
// Input block stage 1: per node (128-thread block):
//   h1 = relu(GN(x @ w1)) -> bf16   and   t = GN(x @ wt) -> fp32
// ---------------------------------------------------------------------------
__global__ __launch_bounds__(128)
void input_proj_kernel(const float* __restrict__ x,
                       const float* __restrict__ w1, const float* __restrict__ g1,
                       const float* __restrict__ b1,
                       const float* __restrict__ wt, const float* __restrict__ gt,
                       const float* __restrict__ bt,
                       __bf16* __restrict__ h1out, float* __restrict__ tout)
{
  __shared__ float xs[22];
  __shared__ float sm[16];
  const int node = blockIdx.x;
  const int c    = threadIdx.x;

  if (c < 22) xs[c] = x[(size_t)node * 22 + c];
  __syncthreads();

  float h = 0.f, t = 0.f;
#pragma unroll
  for (int k = 0; k < 22; ++k) {
    const float xv = xs[k];
    h = fmaf(xv, w1[k * NMAP + c], h);
    t = fmaf(xv, wt[k * NMAP + c], t);
  }

  float sh = h, qh = h * h, st = t, qt = t * t;
#pragma unroll
  for (int o = 16; o; o >>= 1) {
    sh += __shfl_xor(sh, o, 32); qh += __shfl_xor(qh, o, 32);
    st += __shfl_xor(st, o, 32); qt += __shfl_xor(qt, o, 32);
  }
  const int wv = c >> 5;
  if ((c & 31) == 0) { sm[wv] = sh; sm[4 + wv] = qh; sm[8 + wv] = st; sm[12 + wv] = qt; }
  __syncthreads();
  sh = sm[0] + sm[1] + sm[2] + sm[3];
  qh = sm[4] + sm[5] + sm[6] + sm[7];
  st = sm[8] + sm[9] + sm[10] + sm[11];
  qt = sm[12] + sm[13] + sm[14] + sm[15];

  const float muh = sh * (1.f / NMAP), mut = st * (1.f / NMAP);
  const float rh  = rsqrtf(qh * (1.f / NMAP) - muh * muh + GEPS);
  const float rt  = rsqrtf(qt * (1.f / NMAP) - mut * mut + GEPS);

  h1out[(size_t)node * NMAP + c] = (__bf16)fmaxf((h - muh) * rh * g1[c] + b1[c], 0.f);
  tout [(size_t)node * NMAP + c] = (t - mut) * rt * gt[c] + bt[c];
}

// ---------------------------------------------------------------------------
__global__ void zero_kernel(float4* __restrict__ p, int n4)
{
  const int i = blockIdx.x * blockDim.x + threadIdx.x;
  if (i < n4) p[i] = make_float4(0.f, 0.f, 0.f, 0.f);
}

// ---------------------------------------------------------------------------
// Edge gather + scatter-max. ctx stored bf16 (halves gather bytes).
// Thread handles (edge, 16 channels): 32B gather, 16 int atomic-max
// (values >= 0 post-relu so int-bit order == float order; bf16->f32 = bits<<16;
//  agg zero-init matches reference where(isneginf, 0)).
// ---------------------------------------------------------------------------
__global__ __launch_bounds__(256)
void scatter_max_kernel(const unsigned short* __restrict__ ctxb,
                        const int* __restrict__ u, const int* __restrict__ v,
                        float* __restrict__ agg, int E)
{
  const unsigned gid = blockIdx.x * 256u + threadIdx.x;
  const unsigned e   = gid >> 3;
  if (e >= (unsigned)E) return;
  const int part = (int)(gid & 7u) << 4;  // 0..112
  const int un = u[e], vn = v[e];

  const uint4* p = (const uint4*)(ctxb + (size_t)un * NMAP + part);
  const uint4 d0 = p[0];
  const uint4 d1 = p[1];
  int* ag = (int*)(agg + (size_t)vn * NMAP + part);

  const unsigned w[8] = {d0.x, d0.y, d0.z, d0.w, d1.x, d1.y, d1.z, d1.w};
#pragma unroll
  for (int j = 0; j < 8; ++j) {
    atomicMax(ag + 2 * j,     (int)(w[j] << 16));
    atomicMax(ag + 2 * j + 1, (int)(w[j] & 0xFFFF0000u));
  }
}

// ---------------------------------------------------------------------------
extern "C" void kernel_launch(void* const* d_in, const int* in_sizes, int n_in,
                              void* d_out, int out_size, void* d_ws, size_t ws_size,
                              hipStream_t stream)
{
  (void)in_sizes; (void)n_in; (void)out_size; (void)ws_size;

  const float* feats = (const float*)d_in[0];
  const int*   u     = (const int*)d_in[1];
  const int*   v     = (const int*)d_in[2];
  const float* in_w1 = (const float*)d_in[3];
  const float* in_g1 = (const float*)d_in[4];
  const float* in_b1 = (const float*)d_in[5];
  const float* in_w2 = (const float*)d_in[6];
  const float* in_g2 = (const float*)d_in[7];
  const float* in_b2 = (const float*)d_in[8];
  const float* in_wt = (const float*)d_in[9];
  const float* in_gt = (const float*)d_in[10];
  const float* in_bt = (const float*)d_in[11];
  const float* fc1_w = (const float*)d_in[12];
  const float* fc1_g = (const float*)d_in[13];
  const float* fc1_b = (const float*)d_in[14];
  const float* fc2_w = (const float*)d_in[15];
  const float* fc2_g = (const float*)d_in[16];
  const float* fc2_b = (const float*)d_in[17];
  const float* lin_w = (const float*)d_in[18];
  const float* lin_g = (const float*)d_in[19];
  const float* lin_b = (const float*)d_in[20];
  float* out = (float*)d_out;

  const int N = 100000, E = 1000000;
  const size_t SZ  = (size_t)N * NMAP * sizeof(float);   // 51.2 MB
  const size_t SZH = (size_t)N * NMAP * sizeof(__bf16);  // 25.6 MB
  char* ws = (char*)d_ws;
  float*  f0   = (float*)(ws);                       // fp32 feat (residual)
  float*  f1   = (float*)(ws + SZ);
  float*  bufD = (float*)(ws + 2 * SZ);              // t / agg (fp32)
  __bf16* f0b  = (__bf16*)(ws + 3 * SZ);             // bf16 feat
  __bf16* f1b  = (__bf16*)(ws + 3 * SZ + SZH);
  __bf16* ctxb = (__bf16*)(ws + 3 * SZ + 2 * SZH);   // h1 / ctx / h (bf16)
  __bf16* wpb  = (__bf16*)(ws + 3 * SZ + 3 * SZH);   // packed bf16 weights

  // Packed weight slots (elements): K=128 -> 16384, K=256 -> 32768
  __bf16* w2p = wpb;
  __bf16* fc1p = wpb + 16384;                         // 4 x 16384
  __bf16* fc2p = wpb + 16384 * 5;                     // 4 x 32768
  __bf16* linp = wpb + 16384 * 5 + 32768 * 4;         // 4 x 16384

  // ---- pack all weights once per launch
  pack_w_kernel<<<4, 256, 0, stream>>>(in_w2, w2p);
  for (int k = 0; k < 4; ++k) {
    pack_w_kernel<<<4, 256, 0, stream>>>(fc1_w + (size_t)k * NMAP * NMAP, fc1p + (size_t)k * 16384);
    pack_w_kernel<<<8, 256, 0, stream>>>(fc2_w + (size_t)k * 2 * NMAP * NMAP, fc2p + (size_t)k * 32768);
    pack_w_kernel<<<4, 256, 0, stream>>>(lin_w + (size_t)k * NMAP * NMAP, linp + (size_t)k * 16384);
  }

  const int gemm_grid = N / 32;  // 3125 exact

  // ---- input residual block
  input_proj_kernel<<<N, 128, 0, stream>>>(feats, in_w1, in_g1, in_b1,
                                           in_wt, in_gt, in_bt, ctxb, bufD);
  gemm_gn_kernel<4><<<gemm_grid, 256, 0, stream>>>(ctxb, nullptr, w2p, in_g2, in_b2,
                                                   bufD, f0, f0b, 1);

  // ---- 4 aggregation blocks (scale index i = k & 1)
  float*  fin  = f0;
  __bf16* finb = f0b;
  for (int k = 0; k < 4; ++k) {
    const int i = k & 1;

    // ctx = relu(GN(feat @ fc1_w[k]))  -> bf16 only
    gemm_gn_kernel<4><<<gemm_grid, 256, 0, stream>>>(
        finb, nullptr, fc1p + (size_t)k * 16384,
        fc1_g + k * NMAP, fc1_b + k * NMAP, nullptr, nullptr, ctxb, 1);

    // agg = scatter_max(ctx[u], v)
    const int n4 = N * NMAP / 4;
    zero_kernel<<<(n4 + 255) / 256, 256, 0, stream>>>((float4*)bufD, n4);
    const long nt = (long)E * 8;
    scatter_max_kernel<<<(int)((nt + 255) / 256), 256, 0, stream>>>(
        (const unsigned short*)ctxb, u + (size_t)i * E, v + (size_t)i * E, bufD, E);

    // h = relu(GN(concat([feat, agg]) @ fc2_w[k]))  -> bf16 only (reuse ctxb)
    gemm_gn_kernel<8><<<gemm_grid, 256, 0, stream>>>(
        finb, bufD, fc2p + (size_t)k * 32768,
        fc2_g + k * NMAP, fc2_b + k * NMAP, nullptr, nullptr, ctxb, 1);

    // feat = relu(GN(h @ lin_w[k]) + feat)  -> fp32 (+bf16 for next block)
    float*  fout  = (k == 3) ? out : ((fin == f0) ? f1 : f0);
    __bf16* foutb = (finb == f0b) ? f1b : f0b;
    gemm_gn_kernel<4><<<gemm_grid, 256, 0, stream>>>(
        ctxb, nullptr, linp + (size_t)k * 16384,
        lin_g + k * NMAP, lin_b + k * NMAP, fin, fout, foutb, 1);
    fin  = fout;
    finb = foutb;
  }
}
